// MultiMPNN_53240414601509
// MI455X (gfx1250) — matrix operand order repaired and checked by reading, not compiled, not run on hardware
//
#include <hip/hip_runtime.h>

typedef __attribute__((ext_vector_type(16))) _Float16 v16h;
typedef __attribute__((ext_vector_type(8)))  float    v8f;

#define DEV __device__ __forceinline__

static constexpr int NN = 10000;   // nodes
static constexpr int EE = 100000;  // edges
static constexpr int HH = 100;     // hidden
static constexpr int GG = 50000;   // groups
static constexpr int TW = 5;       // towers
static constexpr int MT = 4;       // 16-row M-tiles per wave (64 rows)
static constexpr float EPSF = 1e-5f;

// ---------------- float atomic min/max via ordered-int trick ----------------
DEV void atomicMaxF(float* a, float v) {
  v += 0.0f; // canonicalize -0 -> +0
  if (v >= 0.0f) atomicMax((int*)a, __float_as_int(v));
  else           atomicMin((unsigned int*)a, __float_as_uint(v));
}
DEV void atomicMinF(float* a, float v) {
  v += 0.0f;
  if (v >= 0.0f) atomicMin((int*)a, __float_as_int(v));
  else           atomicMax((unsigned int*)a, __float_as_uint(v));
}

// ---------------- WMMA helpers ----------------
// A tile in LDS: 16x32 f16 row-major As[m*32+k].
// A frag (ISA 7.12.2, 16-bit A 16x32): ka = i + 8*((i>>3) + (lane>>4)), M = lane&15
//   -> per lane two contiguous 16B runs => 2x ds_load_b128.
// B frag (16-bit B 32x16): same K order as A: kb = i + 8*((i>>3) + (lane>>4)); N = lane&15.
//   -> loaded DIRECTLY from global W: per i, half-wave reads 16 consecutive floats (coalesced).
DEV v8f wmma_tile_a(const _Float16* As, v16h bf, int lane, v8f acc) {
  v16h af;
  const int hi = lane >> 4, ln = lane & 15;
#pragma unroll
  for (int i = 0; i < 16; ++i) {
    const int ka = i + ((i >> 3) + hi) * 8;
    af[i] = As[ln * 32 + ka];
  }
  return __builtin_amdgcn_wmma_f32_16x16x32_f16(false, af, false, bf, (short)0, acc, false, false);
}

DEV v16h load_b_frag(const float* __restrict__ Wb, int ldw, int col0, int k0,
                     int Ncols, int K, int lane) {
  v16h bf;
  const int hi = lane >> 4, ln = lane & 15;
  const int gn = col0 + ln;
  if (k0 + 32 <= K && col0 + 16 <= Ncols) {
    const int base = k0 * ldw + gn;   // 32-bit offsets -> saddr addressing
#pragma unroll
    for (int i = 0; i < 16; ++i)
      bf[i] = (_Float16)Wb[base + (i + ((i >> 3) + hi) * 8) * ldw];
  } else {
#pragma unroll
    for (int i = 0; i < 16; ++i) {
      const int gk = k0 + i + ((i >> 3) + hi) * 8;
      float v = 0.0f;
      if (gk < K && gn < Ncols) v = Wb[gk * ldw + gn];
      bf[i] = (_Float16)v;
    }
  }
  return bf;
}

// Load MT consecutive 16x32 A tiles (rows row0..row0+16*MT) into As (MT*512 halves).
template <typename AT>
DEV void load_a_tiles(const AT* __restrict__ Ab, int lda, int row0, int k0,
                      int M, int K, int lane, _Float16* As) {
  if (k0 + 32 <= K && row0 + 16 * MT <= M) {
    if constexpr (sizeof(AT) == 2) {
      // packed 2-half loads; requires even lda & even k0 (true for all f16 operands here)
      const int m0 = lane >> 4, ku = lane & 15;       // j advances m by 2
      const int base = (row0 + m0) * lda + k0 + 2 * ku;
#pragma unroll
      for (int j = 0; j < 8 * MT; ++j) {
        const int idx = lane + j * 32;                // uint index == m*16 + ku
        ((unsigned int*)As)[idx] = *(const unsigned int*)(Ab + base + (j * 2) * lda);
      }
    } else {
      const int m0 = lane >> 5, k = lane & 31;        // j advances m by 1
      const int base = (row0 + m0) * lda + k0 + k;
#pragma unroll
      for (int j = 0; j < 16 * MT; ++j) {
        const int idx = lane + j * 32;                // half index == m*32 + k
        As[idx] = (_Float16)Ab[base + j * lda];
      }
    }
  } else {
    // tail tiles only: keep rolled to limit static code size
#pragma unroll 1
    for (int j = 0; j < 16 * MT; ++j) {
      const int idx = lane + j * 32;
      const int m = idx >> 5, k = idx & 31;
      const int gm = row0 + m, gk = k0 + k;
      float v = 0.0f;
      if (gm < M && gk < K) v = (float)Ab[gm * lda + gk];
      As[idx] = (_Float16)v;
    }
  }
}

// ---------------- generic WMMA GEMM: C = [accum +=alpha*] (A @ W + bias), opt relu ----------------
// One wave per block computes a 64x16 C tile (MT=4 sub-tiles). Batched over gridDim.z.
template <typename AT>
__global__ __launch_bounds__(32)
void mpnn_gemm_wmma(const AT* __restrict__ A, int lda, int abatch,
                    const float* __restrict__ W, int ldw, int wbatch,
                    const float* __restrict__ bias, int bbatch,
                    float* __restrict__ C, int ldc, int cbatch,
                    int M, int Ncols, int K, int relu, int accum, float alpha) {
  __shared__ _Float16 As[MT * 16 * 32];
  const int lane = threadIdx.x;
  const int tn = blockIdx.x, tm = blockIdx.y, z = blockIdx.z;
  const AT* Ab = A + z * abatch;
  const float* Wb = W + z * wbatch;
  float* Cb = C + z * cbatch;
  const float* Bb = bias ? (bias + z * bbatch) : nullptr;
  const int row0 = tm * (16 * MT), col0 = tn * 16;
  const int hi = lane >> 4, ln = lane & 15;
  const v8f vzero = {};
  v8f acc[MT];
#pragma unroll
  for (int s = 0; s < MT; ++s) acc[s] = vzero;

  for (int k0 = 0; k0 < K; k0 += 32) {
    load_a_tiles<AT>(Ab, lda, row0, k0, M, K, lane, As);
    const v16h bf = load_b_frag(Wb, ldw, col0, k0, Ncols, K, lane);
    __syncthreads();
#pragma unroll
    for (int s = 0; s < MT; ++s) acc[s] = wmma_tile_a(As + s * 512, bf, lane, acc[s]);
    __syncthreads();
  }

  const int col = col0 + ln;
  if (col < Ncols) {
    const float bv = Bb ? Bb[col] : 0.0f;
#pragma unroll
    for (int s = 0; s < MT; ++s) {
#pragma unroll
      for (int r = 0; r < 8; ++r) {
        const int row = row0 + s * 16 + r + 8 * hi;
        if (row < M) {
          float v = acc[s][r] + bv;
          if (relu) v = fmaxf(v, 0.0f);
          const int off = row * ldc + col;
          if (accum) Cb[off] += alpha * v;
          else Cb[off] = v;
        }
      }
    }
  }
}

// ---------------- fused pre-tower GEMM + PNA scatter aggregation ----------------
// h[g,z,col] = (h_in[g,:] @ pre_w[z] + pre_b[z][col]) * ts[g]; scatter into node accumulators.
__global__ __launch_bounds__(32)
void mpnn_pre_scatter(const _Float16* __restrict__ A /*G x 300*/,
                      const float* __restrict__ W /*5 x 300 x 100*/,
                      const float* __restrict__ bias /*5 x 100*/,
                      const float* __restrict__ nea /*G x 101, col0 = ts*/,
                      const int* __restrict__ ndst,
                      float* __restrict__ asum, float* __restrict__ asq,
                      float* __restrict__ amin, float* __restrict__ amax,
                      int G) {
  __shared__ _Float16 As[MT * 16 * 32];
  const int lane = threadIdx.x;
  const int tn = blockIdx.x, tm = blockIdx.y, z = blockIdx.z;
  const float* Wz = W + z * (300 * 100);
  const int row0 = tm * (16 * MT), col0 = tn * 16;
  const int hi = lane >> 4, ln = lane & 15;
  const v8f vzero = {};
  v8f acc[MT];
#pragma unroll
  for (int s = 0; s < MT; ++s) acc[s] = vzero;

  for (int k0 = 0; k0 < 300; k0 += 32) {
    load_a_tiles<_Float16>(A, 300, row0, k0, G, 300, lane, As);
    const v16h bf = load_b_frag(Wz, 100, col0, k0, 100, 300, lane);
    __syncthreads();
#pragma unroll
    for (int s = 0; s < MT; ++s) acc[s] = wmma_tile_a(As + s * 512, bf, lane, acc[s]);
    __syncthreads();
  }

  const int col = col0 + ln;
  if (col < 100) {
    const float bv = bias[z * 100 + col];
#pragma unroll
    for (int s = 0; s < MT; ++s) {
#pragma unroll
      for (int r = 0; r < 8; ++r) {
        const int g = row0 + s * 16 + r + 8 * hi;
        if (g < G) {
          const float tsv = nea[g * 101];
          const float v = (acc[s][r] + bv) * tsv;
          const int node = ndst[g];
          const int o = node * 500 + z * 100 + col;
          atomicAdd(asum + o, v);
          atomicAdd(asq + o, v * v);
          atomicMaxF(amax + o, v);
          atomicMinF(amin + o, v);
        }
      }
    }
  }
}

// ---------------- small elementwise / scatter kernels ----------------
__global__ void mpnn_fill_f32(float* p, float v, int n) {
  const int i = blockIdx.x * blockDim.x + threadIdx.x;
  if (i < n) p[i] = v;
}

__global__ void mpnn_ones_col(float* ea, int E) {
  const int e = blockIdx.x * blockDim.x + threadIdx.x;
  if (e < E) ea[e * 101] = 1.0f;
}

__global__ void mpnn_avglog(const int* deg, float* scal, int nb) {
  if (blockIdx.x == 0 && threadIdx.x == 0) {
    float s0 = 0.f, s1 = 0.f;
    for (int i = 0; i < nb; ++i) {
      const float d = (float)deg[i];
      s1 += d * logf((float)i + 1.0f);
      s0 += d;
    }
    scal[0] = s1 / fmaxf(s0, 1.0f);
  }
}

__global__ void mpnn_edge_group_agg(const float* __restrict__ ea, const int* __restrict__ src,
                                    const int* __restrict__ dst, const int* __restrict__ g,
                                    float* cnt, int* isrc, int* idst, float* nea, int E) {
  const int t = blockIdx.x * blockDim.x + threadIdx.x;
  if (t >= E * 101) return;
  const int e = t / 101, c = t % 101;
  const int gr = g[e];
  if (c == 0) {
    atomicAdd(cnt + gr, 1.0f);
    atomicAdd(isrc + gr, src[e]);
    atomicAdd(idst + gr, dst[e]);
    atomicAdd(nea + gr * 101, ea[e * 101]);
  } else {
    atomicAdd(nea + gr * 101 + c, ea[e * 101 + c]);
  }
}

__global__ void mpnn_group_finalize(const float* cnt, const int* isrc, const int* idst,
                                    int* nsrc, int* ndst, float* nea, int G) {
  const int g = blockIdx.x * blockDim.x + threadIdx.x;
  if (g >= G) return;
  const float c = cnt[g];
  const int ci = max((int)c, 1);
  nsrc[g] = isrc[g] / ci;
  ndst[g] = idst[g] / ci;
  nea[g * 101] /= fmaxf(c, 1.0f);
}

__global__ void mpnn_build_hin(const float* __restrict__ x, const int* __restrict__ nsrc,
                               const int* __restrict__ ndst, const float* __restrict__ e_enc,
                               _Float16* __restrict__ hin, int G) {
  const int t = blockIdx.x * blockDim.x + threadIdx.x;
  if (t >= G * 300) return;
  const int g = t / 300, c = t % 300;
  float v;
  if (c < 100)       v = x[ndst[g] * 100 + c];
  else if (c < 200)  v = x[nsrc[g] * 100 + (c - 100)];
  else               v = e_enc[g * 100 + (c - 200)];
  hin[t] = (_Float16)v;
}

__global__ void mpnn_deg_count(const int* ndst, float* deg, int G) {
  const int g = blockIdx.x * blockDim.x + threadIdx.x;
  if (g < G) atomicAdd(deg + ndst[g], 1.0f);
}

__global__ void mpnn_aggs_fin(const float* __restrict__ asum, const float* __restrict__ asq,
                              const float* __restrict__ amin, const float* __restrict__ amax,
                              const float* __restrict__ deg, _Float16* __restrict__ aggs, int n) {
  const int t = blockIdx.x * blockDim.x + threadIdx.x;
  if (t >= n * 500) return;
  const int node = t / 500, f = t % 500;
  const float dg = deg[node];
  const float d = fmaxf(dg, 1.0f);
  const float mean = asum[t] / d;
  const float sq = asq[t] / d;
  const float sd = sqrtf(fmaxf(sq - mean * mean, 0.0f) + EPSF);
  const bool has = dg > 0.0f;
  const float mn = has ? amin[t] : 0.0f;
  const float mx = has ? amax[t] : 0.0f;
  const int tw = f / 100, c = f % 100;
  const int base = node * 2000 + tw * 400 + c;
  aggs[base]       = (_Float16)mean;
  aggs[base + 100] = (_Float16)mn;
  aggs[base + 200] = (_Float16)mx;
  aggs[base + 300] = (_Float16)sd;
}

__global__ void mpnn_s12(const float* deg, const float* scal, float* s1, float* s2, int n) {
  const int i = blockIdx.x * blockDim.x + threadIdx.x;
  if (i >= n) return;
  const float dlog = logf(fmaxf(deg[i], 1.0f) + 1.0f);
  const float a = scal[0];
  s1[i] = dlog / a;
  s2[i] = a / dlog;
}

__global__ void mpnn_post_comb(const float* Yx, const float* Y1, const float* Y2, const float* Y3,
                               const float* s1, const float* s2, const float* pb /*5x20*/,
                               float* out, int n) {
  const int t = blockIdx.x * blockDim.x + threadIdx.x;
  if (t >= n * 100) return;
  const int node = t / 100, f = t % 100;
  out[t] = Yx[t] + Y1[t] + s1[node] * Y2[t] + s2[node] * Y3[t] + pb[f];
}

__global__ void mpnn_bn_stats(const float* __restrict__ conv, float* mu, float* var, int n, int h) {
  const int j = blockIdx.x;
  const int tid = threadIdx.x;
  float s = 0.f, q = 0.f;
  for (int i = tid; i < n; i += blockDim.x) {
    const float v = conv[i * h + j];
    s += v; q += v * v;
  }
  __shared__ float ss[256], qq[256];
  ss[tid] = s; qq[tid] = q;
  __syncthreads();
  for (int st = 128; st > 0; st >>= 1) {
    if (tid < st) { ss[tid] += ss[tid + st]; qq[tid] += qq[tid + st]; }
    __syncthreads();
  }
  if (tid == 0) {
    const float m = ss[0] / n;
    mu[j] = m;
    var[j] = qq[0] / n - m * m;
  }
}

__global__ void mpnn_x_update(float* x, const float* conv, const float* mu, const float* var,
                              const float* gmm, const float* beta, int n, int h) {
  const int t = blockIdx.x * blockDim.x + threadIdx.x;
  if (t >= n * h) return;
  const int j = t % h;
  const float y = (conv[t] - mu[j]) * rsqrtf(var[j] + EPSF) * gmm[j] + beta[j];
  x[t] = (x[t] + fmaxf(y, 0.0f)) * 0.5f;
}

__global__ void mpnn_build_updin(const float* __restrict__ x, const float* __restrict__ nea,
                                 const float* __restrict__ ea, const int* __restrict__ src,
                                 const int* __restrict__ g, _Float16* __restrict__ u, int E) {
  const int t = blockIdx.x * blockDim.x + threadIdx.x;
  if (t >= E * 300) return;
  const int e = t / 300, c = t % 300;
  float v;
  if (c < 100)      v = x[src[e] * 100 + c];
  else if (c < 200) v = nea[g[e] * 101 + 1 + (c - 100)];
  else              v = ea[e * 101 + 1 + (c - 200)];
  u[t] = (_Float16)v;
}

__global__ void mpnn_build_z(const float* __restrict__ x, const float* __restrict__ ea,
                             const int* __restrict__ src, const int* __restrict__ dst,
                             _Float16* __restrict__ z, int E) {
  const int t = blockIdx.x * blockDim.x + threadIdx.x;
  if (t >= E * 300) return;
  const int e = t / 300, c = t % 300;
  float v;
  if (c < 100)      v = fmaxf(x[src[e] * 100 + c], 0.0f);
  else if (c < 200) v = fmaxf(x[dst[e] * 100 + (c - 100)], 0.0f);
  else              v = ea[e * 101 + 1 + (c - 200)];
  z[t] = (_Float16)v;
}

// ============================================================================
extern "C" void kernel_launch(void* const* d_in, const int* in_sizes, int n_in,
                              void* d_out, int out_size, void* d_ws, size_t ws_size,
                              hipStream_t stream) {
  // Input order: setup_inputs() dict insertion order, params flattened recursively
  // in insertion order (layers list in order, 14 leaves per layer).
  const float* x_raw   = (const float*)d_in[0];   // 10000 x 64
  const float* ea_raw  = (const float*)d_in[1];   // 100000 x 32
  const float* node_w  = (const float*)d_in[2];
  const float* node_b  = (const float*)d_in[3];
  const float* edge_w  = (const float*)d_in[4];
  const float* edge_b  = (const float*)d_in[5];
  auto LPm = [&](int l, int j) -> const float* { return (const float*)d_in[6 + 14 * l + j]; };
  const float* mlp_w1 = (const float*)d_in[34];
  const float* mlp_b1 = (const float*)d_in[35];
  const float* mlp_w2 = (const float*)d_in[36];
  const float* mlp_b2 = (const float*)d_in[37];
  const float* mlp_w3 = (const float*)d_in[38];
  const float* mlp_b3 = (const float*)d_in[39];
  const int* eidx     = (const int*)d_in[40];     // 2 x E
  const int* gidx     = (const int*)d_in[41];     // E
  const int* deg_hist = (const int*)d_in[42];     // 32
  const int* srcv = eidx;
  const int* dstv = eidx + EE;
  float* outp = (float*)d_out;

  // ---- workspace carve ----
  char* wsp = (char*)d_ws;
  auto alloc = [&](size_t bytes) -> void* {
    void* p = (void*)wsp;
    wsp += (bytes + 255) & ~(size_t)255;
    return p;
  };
  float* scal   = (float*)alloc(64);
  float* xbuf   = (float*)alloc((size_t)NN * HH * 4);
  float* convb  = (float*)alloc((size_t)NN * HH * 4);
  float* outb   = (float*)alloc((size_t)NN * HH * 4);
  float* bn_mu  = (float*)alloc(HH * 4);
  float* bn_var = (float*)alloc(HH * 4);
  float* edge_attr = (float*)alloc((size_t)EE * 101 * 4);
  float* nea    = (float*)alloc((size_t)GG * 101 * 4);
  float* cnt    = (float*)alloc((size_t)GG * 4);
  int*   isrc   = (int*)alloc((size_t)GG * 4);
  int*   idst   = (int*)alloc((size_t)GG * 4);
  int*   nsrc   = (int*)alloc((size_t)GG * 4);
  int*   ndst   = (int*)alloc((size_t)GG * 4);
  float* e_enc  = (float*)alloc((size_t)GG * 100 * 4);
  _Float16* hin = (_Float16*)alloc((size_t)GG * 300 * 2);
  float* asum   = (float*)alloc((size_t)NN * 500 * 4);
  float* asq    = (float*)alloc((size_t)NN * 500 * 4);
  float* amin   = (float*)alloc((size_t)NN * 500 * 4);
  float* amax   = (float*)alloc((size_t)NN * 500 * 4);
  float* degb   = (float*)alloc((size_t)NN * 4);
  float* s1b    = (float*)alloc((size_t)NN * 4);
  float* s2b    = (float*)alloc((size_t)NN * 4);
  float* Yx     = (float*)alloc((size_t)NN * 100 * 4);
  float* Y1     = (float*)alloc((size_t)NN * 100 * 4);
  float* Y2     = (float*)alloc((size_t)NN * 100 * 4);
  float* Y3     = (float*)alloc((size_t)NN * 100 * 4);
  _Float16* updin = (_Float16*)alloc((size_t)EE * 300 * 2); // also aggs (NN*2000) and z
  float* hmid   = (float*)alloc((size_t)EE * 100 * 4);       // also h1 (E*50), h2 (E*25)
  _Float16* aggs = (_Float16*)updin;
  float* h1 = hmid;
  float* h2 = hmid + (size_t)EE * 50;

  auto fill = [&](float* p, float v, int n) {
    mpnn_fill_f32<<<dim3((unsigned)((n + 255) / 256)), 256, 0, stream>>>(p, v, n);
  };
  auto gemm_f32 = [&](const float* A, int lda, int ab, const float* W, int ldw, int wb,
                      const float* bias, int bb, float* C, int ldc, int cb,
                      int M, int Nc, int K, int relu, int accum, float alpha, int zdim) {
    dim3 g((Nc + 15) / 16, (M + 16 * MT - 1) / (16 * MT), zdim);
    mpnn_gemm_wmma<float><<<g, 32, 0, stream>>>(A, lda, ab, W, ldw, wb, bias, bb, C, ldc, cb,
                                                M, Nc, K, relu, accum, alpha);
  };
  auto gemm_f16 = [&](const _Float16* A, int lda, int ab, const float* W, int ldw, int wb,
                      const float* bias, int bb, float* C, int ldc, int cb,
                      int M, int Nc, int K, int relu, int accum, float alpha, int zdim) {
    dim3 g((Nc + 15) / 16, (M + 16 * MT - 1) / (16 * MT), zdim);
    mpnn_gemm_wmma<_Float16><<<g, 32, 0, stream>>>(A, lda, ab, W, ldw, wb, bias, bb, C, ldc, cb,
                                                   M, Nc, K, relu, accum, alpha);
  };

  const float INFP = __builtin_huge_valf();

  // ---- prologue ----
  mpnn_avglog<<<1, 32, 0, stream>>>(deg_hist, scal, 32);
  // x = x_raw @ node_emb_w + b
  gemm_f32(x_raw, 64, 0, node_w, 100, 0, node_b, 0, xbuf, 100, 0, NN, 100, 64, 0, 0, 0.f, 1);
  // edge_attr = [1, ea_raw @ edge_emb_w + b]
  mpnn_ones_col<<<(EE + 255) / 256, 256, 0, stream>>>(edge_attr, EE);
  gemm_f32(ea_raw, 32, 0, edge_w, 100, 0, edge_b, 0, edge_attr + 1, 101, 0, EE, 100, 32, 0, 0, 0.f, 1);

  for (int l = 0; l < 2; ++l) {
    const float* ee_w  = LPm(l, 0);  const float* ee_b  = LPm(l, 1);
    const float* pre_w = LPm(l, 2);  const float* pre_b = LPm(l, 3);
    const float* post_w = LPm(l, 4); const float* post_b = LPm(l, 5);
    const float* lin_w = LPm(l, 6);  const float* lin_b = LPm(l, 7);
    const float* bn_g  = LPm(l, 8);  const float* bn_b  = LPm(l, 9);
    const float* em_w1 = LPm(l, 10); const float* em_b1 = LPm(l, 11);
    const float* em_w2 = LPm(l, 12); const float* em_b2 = LPm(l, 13);

    // --- group (simplified-edge) aggregation ---
    fill(cnt, 0.f, GG);
    fill((float*)isrc, 0.f, GG);
    fill((float*)idst, 0.f, GG);
    fill(nea, 0.f, GG * 101);
    mpnn_edge_group_agg<<<(EE * 101 + 255) / 256, 256, 0, stream>>>(
        edge_attr, srcv, dstv, gidx, cnt, isrc, idst, nea, EE);
    mpnn_group_finalize<<<(GG + 255) / 256, 256, 0, stream>>>(cnt, isrc, idst, nsrc, ndst, nea, GG);

    // --- e_enc = n_e @ ee_w + ee_b ---
    gemm_f32(nea + 1, 101, 0, ee_w, 100, 0, ee_b, 0, e_enc, 100, 0, GG, 100, 100, 0, 0, 0.f, 1);

    // --- h_in = [x[n_dst], x[n_src], e_enc] (f16) ---
    mpnn_build_hin<<<(GG * 300 + 255) / 256, 256, 0, stream>>>(
        xbuf, nsrc, ndst, e_enc, hin, GG);

    // --- node accumulators + degree ---
    fill(asum, 0.f, NN * 500);
    fill(asq, 0.f, NN * 500);
    fill(amin, INFP, NN * 500);
    fill(amax, -INFP, NN * 500);
    fill(degb, 0.f, NN);
    mpnn_deg_count<<<(GG + 255) / 256, 256, 0, stream>>>(ndst, degb, GG);

    // --- fused per-tower pre-GEMM + ts scale + scatter aggregation ---
    {
      dim3 g((100 + 15) / 16, (GG + 16 * MT - 1) / (16 * MT), TW);
      mpnn_pre_scatter<<<g, 32, 0, stream>>>(hin, pre_w, pre_b, nea, ndst,
                                             asum, asq, amin, amax, GG);
    }

    // --- aggregator finalize + degree scalers ---
    mpnn_aggs_fin<<<(NN * 500 + 255) / 256, 256, 0, stream>>>(
        asum, asq, amin, amax, degb, aggs, NN);
    mpnn_s12<<<(NN + 255) / 256, 256, 0, stream>>>(degb, scal, s1b, s2b, NN);

    // --- post GEMMs (decomposed 1300-wide input), batched over towers ---
    gemm_f32(xbuf, 100, 0, post_w, 20, 26000, nullptr, 0, Yx, 100, 20, NN, 20, 100, 0, 0, 0.f, TW);
    gemm_f16(aggs, 2000, 400, post_w + 100 * 20, 20, 26000, nullptr, 0, Y1, 100, 20, NN, 20, 400, 0, 0, 0.f, TW);
    gemm_f16(aggs, 2000, 400, post_w + 500 * 20, 20, 26000, nullptr, 0, Y2, 100, 20, NN, 20, 400, 0, 0, 0.f, TW);
    gemm_f16(aggs, 2000, 400, post_w + 900 * 20, 20, 26000, nullptr, 0, Y3, 100, 20, NN, 20, 400, 0, 0, 0.f, TW);
    mpnn_post_comb<<<(NN * 100 + 255) / 256, 256, 0, stream>>>(
        Yx, Y1, Y2, Y3, s1b, s2b, post_b, outb, NN);

    // --- conv = out @ lin_w + lin_b ---
    gemm_f32(outb, 100, 0, lin_w, 100, 0, lin_b, 0, convb, 100, 0, NN, 100, 100, 0, 0, 0.f, 1);

    // --- BN + residual ---
    mpnn_bn_stats<<<HH, 256, 0, stream>>>(convb, bn_mu, bn_var, NN, HH);
    mpnn_x_update<<<(NN * HH + 255) / 256, 256, 0, stream>>>(
        xbuf, convb, bn_mu, bn_var, bn_g, bn_b, NN, HH);

    // --- edge MLP update: e += 0.5 * (relu([x[src], n_e[g], e] @ w1 + b1) @ w2 + b2) ---
    mpnn_build_updin<<<(EE / 4 * 300 * 4 + 255) / 256, 256, 0, stream>>>(
        xbuf, nea, edge_attr, srcv, gidx, updin, EE);
    gemm_f16(updin, 300, 0, em_w1, 100, 0, em_b1, 0, hmid, 100, 0, EE, 100, 300, 1, 0, 0.f, 1);
    gemm_f32(hmid, 100, 0, em_w2, 100, 0, em_b2, 0, edge_attr + 1, 101, 0, EE, 100, 100, 0, 1, 0.5f, 1);
  }

  // ---- final edge MLP ----
  _Float16* zb = updin;
  mpnn_build_z<<<(EE / 4 * 300 * 4 + 255) / 256, 256, 0, stream>>>(
      xbuf, edge_attr, srcv, dstv, zb, EE);
  gemm_f16(zb, 300, 0, mlp_w1, 50, 0, mlp_b1, 0, h1, 50, 0, EE, 50, 300, 1, 0, 0.f, 1);
  gemm_f32(h1, 50, 0, mlp_w2, 25, 0, mlp_b2, 0, h2, 25, 0, EE, 25, 50, 1, 0, 0.f, 1);
  gemm_f32(h2, 25, 0, mlp_w3, 2, 0, mlp_b3, 0, outp, 2, 0, EE, 2, 25, 0, 0, 0.f, 1);
}
